// LinearAttention_50740743635564
// MI455X (gfx1250) — matrix operand order repaired and checked by reading, not compiled, not run on hardware
//
#include <hip/hip_runtime.h>

// ---------------------------------------------------------------------------
// MI455X (gfx1250) linear-attention forward, bf16 WMMA everywhere.
// Compute-bound (~480 GFLOP vs ~17us of HBM traffic @23.3TB/s): all matmuls
// use v_wmma_f32_16x16x32_bf16. Big GEMMs stage tiles via the Tensor Data
// Mover (tensor_load_to_lds, TENSORcnt) with double-buffered LDS.
// ---------------------------------------------------------------------------

typedef __attribute__((ext_vector_type(16))) __bf16 bf16x16;
typedef __attribute__((ext_vector_type(8)))  float  f32x8;
typedef __attribute__((ext_vector_type(4)))  unsigned int u32x4;
typedef __attribute__((ext_vector_type(8)))  int  i32x8;
typedef __attribute__((ext_vector_type(4)))  int  i32x4;

#if __has_builtin(__builtin_amdgcn_tensor_load_to_lds) && \
    __has_builtin(__builtin_amdgcn_s_wait_tensorcnt)
#define HAVE_TDM 1
#else
#define HAVE_TDM 0
#endif

union FragU { bf16x16 v; uint4 q[2]; };

// A-fragment (16x32, M x K), row-major source with leading dim ld (elements).
// Per ISA layout: lane m+16g holds K = {8g..8g+7} U {16+8g..16+8g+7}.
__device__ __forceinline__ bf16x16 load_a_frag(const __bf16* base, int ld) {
    int lane = threadIdx.x & 31;
    int m = lane & 15, g = lane >> 4;
    const __bf16* p = base + (size_t)m * ld;
    FragU f;
    f.q[0] = *(const uint4*)(p + 8 * g);
    f.q[1] = *(const uint4*)(p + 16 + 8 * g);
    return f.v;
}

// B-fragment (32x16, K x N) loaded from B^T stored row-major (ld = elements):
// lane n+16h holds column n's K = {8h..8h+7} U {16+8h..16+8h+7} (B^T row n).
__device__ __forceinline__ bf16x16 load_b_frag(const __bf16* base, int ld) {
    int lane = threadIdx.x & 31;
    int nn = lane & 15, hh = lane >> 4;
    const __bf16* p = base + (size_t)nn * ld;
    FragU f;
    f.q[0] = *(const uint4*)(p + 8 * hh);
    f.q[1] = *(const uint4*)(p + 16 + 8 * hh);
    return f.v;
}

__device__ __forceinline__ f32x8 wmma_bf(bf16x16 a, bf16x16 b, f32x8 c) {
    return __builtin_amdgcn_wmma_f32_16x16x32_bf16(false, a, false, b,
                                                   (short)0, c, false, false);
}

#if HAVE_TDM
// TDM 2-D tile load: 128 rows x 32 bf16 elements, row stride = strideElems,
// packed contiguously into LDS (rows back-to-back = [128][32] layout).
// D# layout per CDNA5 ISA.
__device__ __forceinline__ void tdm_load_128x32(const __bf16* g, __bf16* lds,
                                                unsigned strideElems) {
    unsigned long long ga = (unsigned long long)(size_t)g;
    unsigned ldsa = (unsigned)(size_t)lds;          // low 32 bits = LDS offset
    u32x4 g0;
    g0[0] = 1u;                                     // count=1, user mode
    g0[1] = ldsa;                                   // lds_addr
    g0[2] = (unsigned)ga;                           // global_addr[31:0]
    g0[3] = (unsigned)((ga >> 32) & 0x01ffffffu)    // global_addr[56:32]
            | (2u << 30);                           // type = 2 ("image")
    const unsigned d0 = 32, d1 = 128, t0 = 32, t1 = 128;
    unsigned w0 = (1u << 16);                       // data_size=1 -> 2 bytes
    unsigned w1 = (d0 & 0xffffu) << 16;             // tensor_dim0[15:0] @ 63:48
    unsigned w2 = ((d0 >> 16) & 0xffffu)            // tensor_dim0[31:16]
                | ((d1 & 0xffffu) << 16);           // tensor_dim1[15:0]
    unsigned w3 = ((d1 >> 16) & 0xffffu)            // tensor_dim1[31:16]
                | (t0 << 16);                       // tile_dim0
    unsigned w4 = t1;                               // tile_dim1 (tile_dim2=0)
    unsigned w5 = strideElems;                      // tensor_dim0_stride[31:0]
    i32x8 g1 = {(int)w0, (int)w1, (int)w2, (int)w3, (int)w4, (int)w5, 0, 0};
    i32x4 z4 = {0, 0, 0, 0};
#if defined(__clang_major__) && (__clang_major__ >= 23)
    i32x8 z8 = {0, 0, 0, 0, 0, 0, 0, 0};
    __builtin_amdgcn_tensor_load_to_lds(g0, g1, z4, z4, z8, 0);
#else
    __builtin_amdgcn_tensor_load_to_lds(g0, g1, z4, z4, 0);
#endif
}
#endif

// ---------------------------------------------------------------------------
// fp32 -> bf16 conversion
// ---------------------------------------------------------------------------
__global__ void cvt_bf16_kernel(const float* __restrict__ src,
                                __bf16* __restrict__ dst, int n) {
    int i = blockIdx.x * blockDim.x + threadIdx.x;
    if (i < n) dst[i] = (__bf16)src[i];
}

// ---------------------------------------------------------------------------
// Generic GEMM: C[M,N] = A[M,K] * Bt[N,K]^T, bf16 in, bf16 or f32 out.
// Block tile 128x128, 8 waves (4M x 2N), wave tile 32x64, K-step 32.
// Tiles staged by TDM (double buffered, TENSORcnt-gated) when available.
// ---------------------------------------------------------------------------
template <bool F32OUT>
__global__ void __launch_bounds__(256)
gemm_bf16_kernel(const __bf16* __restrict__ A, const __bf16* __restrict__ Bt,
                 void* __restrict__ C, int M, int N, int K) {
    __shared__ __bf16 As[2][128][32];
    __shared__ __bf16 Bs[2][128][32];
    int tid = threadIdx.x;
    int wave = tid >> 5, lane = tid & 31;
    int wm = wave & 3, wn = wave >> 2;
    int row0 = blockIdx.y * 128;
    int col0 = blockIdx.x * 128;
    int nk = K >> 5;

    f32x8 acc[2][4] = {};

#if HAVE_TDM
    if (wave == 0) {   // TDM issues once per wave; wave 0 drives the DMA
        tdm_load_128x32(&A[(size_t)row0 * K], &As[0][0][0], (unsigned)K);
        tdm_load_128x32(&Bt[(size_t)col0 * K], &Bs[0][0][0], (unsigned)K);
    }
    for (int kk = 0; kk < nk; ++kk) {
        int cur = kk & 1;
        if (wave == 0) __builtin_amdgcn_s_wait_tensorcnt(0);
        __syncthreads();                      // stage `cur` visible to all
        if (wave == 0 && kk + 1 < nk) {       // prefetch next stage via TDM
            tdm_load_128x32(&A[(size_t)row0 * K + (kk + 1) * 32],
                            &As[cur ^ 1][0][0], (unsigned)K);
            tdm_load_128x32(&Bt[(size_t)col0 * K + (kk + 1) * 32],
                            &Bs[cur ^ 1][0][0], (unsigned)K);
        }
        bf16x16 a0 = load_a_frag(&As[cur][wm * 32][0], 32);
        bf16x16 a1 = load_a_frag(&As[cur][wm * 32 + 16][0], 32);
#pragma unroll
        for (int t = 0; t < 4; ++t) {
            bf16x16 b = load_b_frag(&Bs[cur][wn * 64 + t * 16][0], 32);
            acc[0][t] = wmma_bf(a0, b, acc[0][t]);
            acc[1][t] = wmma_bf(a1, b, acc[1][t]);
        }
        __syncthreads();
    }
#else
    for (int kk = 0; kk < nk; ++kk) {
        int k0 = kk * 32;
        {   // stage A/B tiles: 128 rows x 32, two b128 per thread each
            int r = tid >> 1, c = (tid & 1) * 16;
            const __bf16* pa = &A[(size_t)(row0 + r) * K + k0 + c];
            *(uint4*)&As[0][r][c]     = *(const uint4*)(pa);
            *(uint4*)&As[0][r][c + 8] = *(const uint4*)(pa + 8);
            const __bf16* pb = &Bt[(size_t)(col0 + r) * K + k0 + c];
            *(uint4*)&Bs[0][r][c]     = *(const uint4*)(pb);
            *(uint4*)&Bs[0][r][c + 8] = *(const uint4*)(pb + 8);
        }
        __syncthreads();
        bf16x16 a0 = load_a_frag(&As[0][wm * 32][0], 32);
        bf16x16 a1 = load_a_frag(&As[0][wm * 32 + 16][0], 32);
#pragma unroll
        for (int t = 0; t < 4; ++t) {
            bf16x16 b = load_b_frag(&Bs[0][wn * 64 + t * 16][0], 32);
            acc[0][t] = wmma_bf(a0, b, acc[0][t]);
            acc[1][t] = wmma_bf(a1, b, acc[1][t]);
        }
        __syncthreads();
    }
#endif
    // store: lane l -> row = 8*(l>>4)+r, col = l&15
    int rb = row0 + wm * 32 + 8 * (lane >> 4);
    int cn = lane & 15;
#pragma unroll
    for (int mi = 0; mi < 2; ++mi) {
#pragma unroll
        for (int t = 0; t < 4; ++t) {
            int oc = col0 + wn * 64 + t * 16 + cn;
#pragma unroll
            for (int r = 0; r < 8; ++r) {
                float v = acc[mi][t][r];
                if (F32OUT)
                    ((float*)C)[(size_t)(rb + mi * 16 + r) * N + oc] = v;
                else
                    ((__bf16*)C)[(size_t)(rb + mi * 16 + r) * N + oc] = (__bf16)v;
            }
        }
    }
}

// ---------------------------------------------------------------------------
// Feature map: out = (Rh @ f1^T + b1) * (Rh @ f2^T + b2) * scale, per head.
// R is [B*L, D] bf16; head h = columns h*128..h*128+127.
// Writes feat [B,H,L,dk]; optionally transposed copy tpose [B,H,dk,L].
// Grid: (B*L/64, H). Block 256.
// ---------------------------------------------------------------------------
__global__ void __launch_bounds__(256)
featmap_kernel(const __bf16* __restrict__ R,
               const __bf16* __restrict__ f1t, const __bf16* __restrict__ f2t,
               const float* __restrict__ b1, const float* __restrict__ b2,
               float scale, __bf16* __restrict__ feat, __bf16* __restrict__ tpose) {
    const int D = 2048, dk = 128, L = 4096, Hh = 16;
    __shared__ __bf16 Rs[64][128];
    int h = blockIdx.y;
    int row0 = blockIdx.x * 64;  // over B*L
    int tid = threadIdx.x, wave = tid >> 5, lane = tid & 31;
    int wm = wave & 3, wn = wave >> 2;

    {   // stage 64x128 head slice
        int r = tid >> 2, c0 = (tid & 3) * 32;
        const __bf16* src = &R[(size_t)(row0 + r) * D + h * dk + c0];
#pragma unroll
        for (int i = 0; i < 4; ++i)
            *(uint4*)&Rs[r][c0 + i * 8] = *(const uint4*)(src + i * 8);
    }
    __syncthreads();

    f32x8 acc1[4] = {}, acc2[4] = {};
#pragma unroll
    for (int ks = 0; ks < 4; ++ks) {
        bf16x16 a = load_a_frag(&Rs[wm * 16][ks * 32], 128);
#pragma unroll
        for (int t = 0; t < 4; ++t) {
            bf16x16 bb1 = load_b_frag(f1t + (size_t)(wn * 64 + t * 16) * dk + ks * 32, dk);
            acc1[t] = wmma_bf(a, bb1, acc1[t]);
            bf16x16 bb2 = load_b_frag(f2t + (size_t)(wn * 64 + t * 16) * dk + ks * 32, dk);
            acc2[t] = wmma_bf(a, bb2, acc2[t]);
        }
    }
    int hl = lane >> 4, cn = lane & 15;
    int rbase = wm * 16 + 8 * hl;
#pragma unroll
    for (int t = 0; t < 4; ++t) {
        int j = wn * 64 + t * 16 + cn;
        float bb1 = b1[j], bb2 = b2[j];
#pragma unroll
        for (int r = 0; r < 8; ++r) {
            int mm = row0 + rbase + r;       // global row in B*L
            int b = mm / L, l = mm % L;
            float v = (acc1[t][r] + bb1) * (acc2[t][r] + bb2) * scale;
            __bf16 bv = (__bf16)v;
            feat[(((size_t)b * Hh + h) * L + l) * dk + j] = bv;
            if (tpose)
                tpose[(((size_t)b * Hh + h) * dk + j) * L + l] = bv;
        }
    }
}

// ---------------------------------------------------------------------------
// v transpose: Vr [B*L, D] -> v_t [B,H,dk,L]. Grid (L/32, dk/32, B*H), blk(32,8)
// ---------------------------------------------------------------------------
__global__ void vtrans_kernel(const __bf16* __restrict__ Vr,
                              __bf16* __restrict__ v_t) {
    const int D = 2048, L = 4096;
    __shared__ __bf16 tile[32][33];
    int bh = blockIdx.z, b = bh >> 4, h = bh & 15;
    int l0 = blockIdx.x * 32, j0 = blockIdx.y * 32;
#pragma unroll
    for (int i = 0; i < 4; ++i) {
        int l = threadIdx.y + i * 8, j = threadIdx.x;
        tile[l][j] = Vr[(size_t)(b * L + l0 + l) * D + h * 128 + j0 + j];
    }
    __syncthreads();
#pragma unroll
    for (int i = 0; i < 4; ++i) {
        int j = threadIdx.y + i * 8, l = threadIdx.x;
        v_t[((size_t)bh * 128 + j0 + j) * L + l0 + l] = tile[l][j];
    }
}

// ---------------------------------------------------------------------------
// Pass A: per-chunk KV^T[n] = vc^T @ kc  (dv x dk, f32).
// A = v_t rows (dv x C), B^T = k_t rows (dk x C). Grid (n, B*H), 8 waves,
// wave tile 16x128 (8 tiles), K = 64 (2 steps).
// ---------------------------------------------------------------------------
__global__ void __launch_bounds__(256)
chunk_kvt_kernel(const __bf16* __restrict__ v_t, const __bf16* __restrict__ k_t,
                 float* __restrict__ kvt) {
    const int L = 4096, dk = 128;
    int n = blockIdx.x, bh = blockIdx.y;
    int tid = threadIdx.x, wave = tid >> 5, lane = tid & 31;
    const __bf16* va = v_t + ((size_t)bh * dk + wave * 16) * L + n * 64;
    const __bf16* kb = k_t + ((size_t)bh * dk) * L + n * 64;

    f32x8 acc[8] = {};
#pragma unroll
    for (int ks = 0; ks < 2; ++ks) {
        bf16x16 a = load_a_frag(va + ks * 32, L);
#pragma unroll
        for (int t = 0; t < 8; ++t) {
            bf16x16 b = load_b_frag(kb + (size_t)(t * 16) * L + ks * 32, L);
            acc[t] = wmma_bf(a, b, acc[t]);
        }
    }
    float* out = kvt + ((size_t)bh * 64 + n) * (128 * 128);
    int rbase = wave * 16 + 8 * (lane >> 4), cn = lane & 15;
#pragma unroll
    for (int t = 0; t < 8; ++t)
#pragma unroll
        for (int r = 0; r < 8; ++r)
            out[(size_t)(rbase + r) * 128 + t * 16 + cn] = acc[t][r];
}

// ---------------------------------------------------------------------------
// Pass B: exclusive prefix-sum over chunks, f32 accum -> bf16 state.
// Grid (B*H, 64), block 256; each thread owns one (j,d) element.
// ---------------------------------------------------------------------------
__global__ void __launch_bounds__(256)
scan_kernel(const float* __restrict__ kvt, __bf16* __restrict__ kvs) {
    int bh = blockIdx.x;
    int e = blockIdx.y * 256 + threadIdx.x;          // 0..16383
    const float* src = kvt + (size_t)bh * 64 * 16384 + e;
    __bf16* dst = kvs + (size_t)bh * 64 * 16384 + e;
    float acc = 0.f;
    for (int n = 0; n < 64; ++n) {
        dst[(size_t)n * 16384] = (__bf16)acc;        // exclusive
        acc += src[(size_t)n * 16384];
    }
}

// ---------------------------------------------------------------------------
// Pass C: per (b,h,chunk): s = qc@kc^T (masked), o = qc@KV + s@vc, RMSNorm.
// Grid (n, B*H), 8 waves (4M x 2N), out tile 64(c) x 128(j).
// ---------------------------------------------------------------------------
__global__ void __launch_bounds__(256)
attn_out_kernel(const __bf16* __restrict__ q_feat, const __bf16* __restrict__ k_feat,
                const __bf16* __restrict__ v_t, const __bf16* __restrict__ kvs,
                const float* __restrict__ norm_w, __bf16* __restrict__ o_bf) {
    const int L = 4096, dk = 128, D = 2048;
    __shared__ __bf16 s_lds[64][64];
    __shared__ float o_lds[64][128];
    __shared__ float ssq[64];
    int n = blockIdx.x, bh = blockIdx.y, b = bh >> 4, h = bh & 15;
    int tid = threadIdx.x, wave = tid >> 5, lane = tid & 31;
    int wm = wave & 3, wn = wave >> 2;
    int hl = lane >> 4, cn = lane & 15;
    int rbase = wm * 16 + 8 * hl;

    const __bf16* qb = q_feat + ((size_t)bh * L + n * 64) * dk;   // [64,128]
    const __bf16* kb = k_feat + ((size_t)bh * L + n * 64) * dk;   // [64,128]

    // ---- s = qc @ kc^T : 64x64, K=128; wave rows wm*16, cols wn*32 (2 tiles)
    f32x8 sacc[2] = {};
#pragma unroll
    for (int ks = 0; ks < 4; ++ks) {
        bf16x16 a = load_a_frag(qb + (size_t)(wm * 16) * dk + ks * 32, dk);
#pragma unroll
        for (int t = 0; t < 2; ++t) {
            bf16x16 bfr = load_b_frag(kb + (size_t)(wn * 32 + t * 16) * dk + ks * 32, dk);
            sacc[t] = wmma_bf(a, bfr, sacc[t]);
        }
    }
    // causal mask + bf16 into LDS
#pragma unroll
    for (int t = 0; t < 2; ++t) {
        int col = wn * 32 + t * 16 + cn;
#pragma unroll
        for (int r = 0; r < 8; ++r) {
            int row = rbase + r;
            float v = (col <= row) ? sacc[t][r] : 0.f;
            s_lds[row][col] = (__bf16)v;
        }
    }
    __syncthreads();

    // ---- out = qc @ KVS  +  s @ vc : 64x128; wave rows wm*16, cols wn*64
    f32x8 acc[4] = {};
    const __bf16* kvb = kvs + ((size_t)bh * 64 + n) * (128 * 128);   // [j][d]
    const __bf16* vtb = v_t + (size_t)bh * dk * L + n * 64;          // row j: +j*L
#pragma unroll
    for (int ks = 0; ks < 4; ++ks) {   // inter, K = dk = 128
        bf16x16 a = load_a_frag(qb + (size_t)(wm * 16) * dk + ks * 32, dk);
#pragma unroll
        for (int t = 0; t < 4; ++t) {
            bf16x16 bfr = load_b_frag(kvb + (size_t)(wn * 64 + t * 16) * dk + ks * 32, dk);
            acc[t] = wmma_bf(a, bfr, acc[t]);
        }
    }
#pragma unroll
    for (int ks = 0; ks < 2; ++ks) {   // intra, K = C = 64
        bf16x16 a = load_a_frag(&s_lds[wm * 16][ks * 32], 64);
#pragma unroll
        for (int t = 0; t < 4; ++t) {
            bf16x16 bfr = load_b_frag(vtb + (size_t)(wn * 64 + t * 16) * L + ks * 32, L);
            acc[t] = wmma_bf(a, bfr, acc[t]);
        }
    }
    // park in LDS for RMSNorm
#pragma unroll
    for (int t = 0; t < 4; ++t)
#pragma unroll
        for (int r = 0; r < 8; ++r)
            o_lds[rbase + r][wn * 64 + t * 16 + cn] = acc[t][r];
    if (tid < 64) ssq[tid] = 0.f;
    __syncthreads();
    {
        int row = tid >> 2, seg = tid & 3;
        float s = 0.f;
#pragma unroll
        for (int i = 0; i < 32; ++i) {
            float v = o_lds[row][seg * 32 + i];
            s += v * v;
        }
        atomicAdd(&ssq[row], s);
    }
    __syncthreads();
    for (int i = tid; i < 64 * 128; i += 256) {
        int c = i >> 7, j = i & 127;
        float inv = rsqrtf(ssq[c] * (1.0f / 128.0f) + 1e-5f);
        float v = o_lds[c][j] * inv * norm_w[j];
        int l = n * 64 + c;
        o_bf[(size_t)(b * L + l) * D + h * 128 + j] = (__bf16)v;
    }
}

// ---------------------------------------------------------------------------
// Launch
// ---------------------------------------------------------------------------
extern "C" void kernel_launch(void* const* d_in, const int* in_sizes, int n_in,
                              void* d_out, int out_size, void* d_ws, size_t ws_size,
                              hipStream_t stream) {
    (void)in_sizes; (void)n_in; (void)out_size; (void)ws_size;
    const int B = 2, L = 4096, D = 2048, Hh = 16, dk = 128, NC = 64;
    const size_t BL = (size_t)B * L;

    const float* x    = (const float*)d_in[0];
    const float* wq   = (const float*)d_in[1];
    const float* wk   = (const float*)d_in[2];
    const float* wv   = (const float*)d_in[3];
    const float* wo   = (const float*)d_in[4];
    const float* fq1w = (const float*)d_in[5];
    const float* fq1b = (const float*)d_in[6];
    const float* fq2w = (const float*)d_in[7];
    const float* fq2b = (const float*)d_in[8];
    const float* fk1w = (const float*)d_in[9];
    const float* fk1b = (const float*)d_in[10];
    const float* fk2w = (const float*)d_in[11];
    const float* fk2b = (const float*)d_in[12];
    const float* normw = (const float*)d_in[13];

    char* ws = (char*)d_ws;
    size_t off = 0;
    auto alloc = [&](size_t bytes) {
        void* p = ws + off;
        off += (bytes + 255) & ~(size_t)255;
        return p;
    };
    const size_t WB = (size_t)D * D * 2;            // 8 MiB bf16 weight
    __bf16* wq_bf = (__bf16*)alloc(WB);
    __bf16* wk_bf = (__bf16*)alloc(WB);
    __bf16* wv_bf = (__bf16*)alloc(WB);
    __bf16* wo_bf = (__bf16*)alloc(WB);
    const size_t FB = (size_t)dk * dk * 2;
    __bf16* fq1 = (__bf16*)alloc(FB);
    __bf16* fq2 = (__bf16*)alloc(FB);
    __bf16* fk1 = (__bf16*)alloc(FB);
    __bf16* fk2 = (__bf16*)alloc(FB);
    // Region R1 (128 MiB): x_bf + Qr + Kr + Vr; later aliased by kvt, then o_bf
    const size_t TB = BL * D * 2;                   // 32 MiB bf16 tensor
    char* R1 = (char*)alloc(4 * TB);
    __bf16* x_bf = (__bf16*)(R1);
    __bf16* Qr   = (__bf16*)(R1 + TB);
    __bf16* Kr   = (__bf16*)(R1 + 2 * TB);
    __bf16* Vr   = (__bf16*)(R1 + 3 * TB);
    float*  kvt  = (float*)R1;                      // 128 MiB, after QKV dead
    __bf16* o_bf = (__bf16*)R1;                     // 32 MiB, after kvt dead
    __bf16* q_feat = (__bf16*)alloc(TB);
    __bf16* k_feat = (__bf16*)alloc(TB);
    __bf16* k_t    = (__bf16*)alloc(TB);
    __bf16* v_t    = (__bf16*)alloc(TB);
    __bf16* kvs    = (__bf16*)alloc((size_t)B * Hh * NC * dk * dk * 2);  // 64 MiB

    // 1) convert to bf16
    auto cvt = [&](const float* s, __bf16* d, size_t n) {
        cvt_bf16_kernel<<<dim3((unsigned)((n + 255) / 256)), 256, 0, stream>>>(s, d, (int)n);
    };
    cvt(wq, wq_bf, (size_t)D * D);
    cvt(wk, wk_bf, (size_t)D * D);
    cvt(wv, wv_bf, (size_t)D * D);
    cvt(wo, wo_bf, (size_t)D * D);
    cvt(fq1w, fq1, (size_t)dk * dk);
    cvt(fq2w, fq2, (size_t)dk * dk);
    cvt(fk1w, fk1, (size_t)dk * dk);
    cvt(fk2w, fk2, (size_t)dk * dk);
    cvt(x, x_bf, BL * D);

    // 2) QKV projections (TDM-staged, WMMA bf16)
    dim3 ggrid(D / 128, (unsigned)(BL / 128));
    gemm_bf16_kernel<false><<<ggrid, 256, 0, stream>>>(x_bf, wq_bf, Qr, (int)BL, D, D);
    gemm_bf16_kernel<false><<<ggrid, 256, 0, stream>>>(x_bf, wk_bf, Kr, (int)BL, D, D);
    gemm_bf16_kernel<false><<<ggrid, 256, 0, stream>>>(x_bf, wv_bf, Vr, (int)BL, D, D);

    // 3) feature maps (+ head-major reorg; k also transposed)
    const float scale = 0.08838834764831845f;       // 128^-0.5
    dim3 fgrid((unsigned)(BL / 64), Hh);
    featmap_kernel<<<fgrid, 256, 0, stream>>>(Qr, fq1, fq2, fq1b, fq2b, scale,
                                              q_feat, (__bf16*)nullptr);
    featmap_kernel<<<fgrid, 256, 0, stream>>>(Kr, fk1, fk2, fk1b, fk2b, 1.0f,
                                              k_feat, k_t);
    vtrans_kernel<<<dim3(L / 32, dk / 32, B * Hh), dim3(32, 8), 0, stream>>>(Vr, v_t);

    // 4) per-chunk KV^T, 5) exclusive scan, 6) attention + RMSNorm
    chunk_kvt_kernel<<<dim3(NC, B * Hh), 256, 0, stream>>>(v_t, k_t, kvt);
    scan_kernel<<<dim3(B * Hh, 64), 256, 0, stream>>>(kvt, kvs);
    attn_out_kernel<<<dim3(NC, B * Hh), 256, 0, stream>>>(q_feat, k_feat, v_t,
                                                          kvs, normw, o_bf);

    // 7) output projection (f32 out)
    gemm_bf16_kernel<true><<<ggrid, 256, 0, stream>>>(o_bf, wo_bf, d_out, (int)BL, D, D);
}
